// SeparateLinear_29265907155013
// MI455X (gfx1250) — compile-verified, matrix operand order checked by reading, not run on hardware
//
#include <hip/hip_runtime.h>

// ---------------------------------------------------------------------------
// out[B,64] = x[B,1024] @ W[64,1024]^T + b   (fp32 in/out)
// Memory-bound: x stream 256MB + out 16MB -> ~11.7us @ 23.3 TB/s.
// V_WMMA_F32_16X16X32_BF16 with split-precision (hi/lo bf16, truncation split):
//   acc += Ahi*Bhi + Ahi*Blo + Alo*Bhi   (f32 accumulate, ~1e-6 rel err @ K=1024)
// W pre-split to bf16 hi/lo planes in LDS (once per workgroup, ~258KB).
// x loads / out stores are non-temporal so the one-shot streams don't evict
// the L2-resident W.
// ---------------------------------------------------------------------------

#define IN_CH   1024
#define OUT_CH  64
#define BATCH   65536
#define PITCH   (IN_CH + 8)          // bf16 elems; 2064B row stride (16B aligned, bank-skewed)
#define LDS_U16 (2 * OUT_CH * PITCH) // hi plane + lo plane

typedef __bf16          v16bf __attribute__((ext_vector_type(16)));
typedef float           v8f   __attribute__((ext_vector_type(8)));
typedef float           v4f   __attribute__((ext_vector_type(4)));
typedef unsigned short  v16u  __attribute__((ext_vector_type(16)));
typedef unsigned short  v8u   __attribute__((ext_vector_type(8)));
typedef unsigned int    v8ui  __attribute__((ext_vector_type(8)));

extern __shared__ unsigned short smem[]; // [2][OUT_CH][PITCH] : hi plane, lo plane

// Pack the two high-16 halves of (f0,f1) into one dword: {bf16(f1),bf16(f0)}.
// Single v_perm_b32: S0 = u1 (bytes 4-7), S1 = u0 (bytes 0-3).
__device__ __forceinline__ unsigned pack_hi16(unsigned u0, unsigned u1) {
    return __builtin_amdgcn_perm(u1, u0, 0x07060302u);
}

// Truncation split of a float pair -> packed bf16 hi pair + packed bf16 lo pair.
__device__ __forceinline__ void split2(float f0, float f1, unsigned& hi, unsigned& lo) {
    unsigned u0 = __builtin_bit_cast(unsigned, f0);
    unsigned u1 = __builtin_bit_cast(unsigned, f1);
    hi = pack_hi16(u0, u1);                                    // 1x v_perm
    float l0 = f0 - __builtin_bit_cast(float, u0 & 0xFFFF0000u);
    float l1 = f1 - __builtin_bit_cast(float, u1 & 0xFFFF0000u);
    lo = pack_hi16(__builtin_bit_cast(unsigned, l0),
                   __builtin_bit_cast(unsigned, l1));          // 1x v_perm
}

struct Frag { v8ui hi; v8ui lo; };

// A-matrix 16x32 bf16 fragment (ISA layout): lane L holds row M=L%16,
// K runs [kb..kb+7] (VGPR0-3) and [kb+16..kb+23] (VGPR4-7), kb = k0 + (L/16)*8.
__device__ __forceinline__ void load_a_frag(const float* __restrict__ xrow, int kb, Frag& f) {
    const v4f* p0 = reinterpret_cast<const v4f*>(xrow + kb);
    const v4f* p1 = reinterpret_cast<const v4f*>(xrow + kb + 16);
    v4f r0 = __builtin_nontemporal_load(p0);
    v4f r1 = __builtin_nontemporal_load(p0 + 1);
    v4f r2 = __builtin_nontemporal_load(p1);
    v4f r3 = __builtin_nontemporal_load(p1 + 1);
    float v[16] = { r0[0], r0[1], r0[2], r0[3],  r1[0], r1[1], r1[2], r1[3],
                    r2[0], r2[1], r2[2], r2[3],  r3[0], r3[1], r3[2], r3[3] };
#pragma unroll
    for (int j = 0; j < 8; ++j) {
        unsigned h, l;
        split2(v[2 * j], v[2 * j + 1], h, l);
        f.hi[j] = h;
        f.lo[j] = l;
    }
}

// B-matrix 32x16 bf16 fragment from LDS: lane L holds column N (= W row) ch,
// same per-lane K pattern as A. Two 16B ds_load_b128 per plane.
__device__ __forceinline__ v16u load_b_frag(int plane, int ch, int kb) {
    const unsigned short* row = &smem[plane * (OUT_CH * PITCH) + ch * PITCH];
    v8u a = *reinterpret_cast<const v8u*>(row + kb);
    v8u c = *reinterpret_cast<const v8u*>(row + kb + 16);
    return __builtin_shufflevector(a, c, 0, 1, 2, 3, 4, 5, 6, 7,
                                         8, 9, 10, 11, 12, 13, 14, 15);
}

__device__ __forceinline__ v8f wmma_bf16(const v8ui& a, const v16u& b, v8f c) {
    return __builtin_amdgcn_wmma_f32_16x16x32_bf16(
        false, __builtin_bit_cast(v16bf, a),
        false, __builtin_bit_cast(v16bf, b),
        (short)0, c, false, false);
}

__global__ __launch_bounds__(256) void seplinear_wmma_kernel(
    const float* __restrict__ x, const float* __restrict__ W,
    const float* __restrict__ b, float* __restrict__ out) {

    const int tid = threadIdx.x;

    // ---- Stage W into LDS as bf16 hi/lo planes (coalesced, once per WG) ----
    for (int idx = tid; idx < OUT_CH * IN_CH; idx += 256) {
        int ch = idx >> 10, k = idx & (IN_CH - 1);
        float f = W[idx];
        unsigned u = __builtin_bit_cast(unsigned, f);
        float fl = f - __builtin_bit_cast(float, u & 0xFFFF0000u);
        smem[ch * PITCH + k]                  = (unsigned short)(u >> 16);
        smem[OUT_CH * PITCH + ch * PITCH + k] =
            (unsigned short)(__builtin_bit_cast(unsigned, fl) >> 16);
    }
    __syncthreads();

    const int lane  = tid & 31;
    const int wave  = tid >> 5;
    const int lrow  = lane & 15;   // M row within tile (A) / N col (B, C/D)
    const int khalf = lane >> 4;   // 0 or 1

    // Each wave: 32 rows (2 M-tiles) x all 64 outputs (4 N-tiles).
    for (int base = blockIdx.x * 256; base < BATCH; base += gridDim.x * 256) {
        const int m0 = base + wave * 32;

        v8f acc[2][4];
#pragma unroll
        for (int mt = 0; mt < 2; ++mt)
#pragma unroll
            for (int nt = 0; nt < 4; ++nt) acc[mt][nt] = v8f{};

        const float* xr0 = x + (size_t)(m0 + lrow) * IN_CH;
        const float* xr1 = x + (size_t)(m0 + 16 + lrow) * IN_CH;

        for (int k0 = 0; k0 < IN_CH; k0 += 32) {
            const int kb = k0 + khalf * 8;
            Frag a0, a1;
            load_a_frag(xr0, kb, a0);
            load_a_frag(xr1, kb, a1);

#pragma unroll
            for (int nt = 0; nt < 4; ++nt) {
                const int ch = nt * 16 + lrow;
                v16u bh = load_b_frag(0, ch, kb);
                v16u bl = load_b_frag(1, ch, kb);

                acc[0][nt] = wmma_bf16(a0.hi, bh, acc[0][nt]);
                acc[0][nt] = wmma_bf16(a0.hi, bl, acc[0][nt]);
                acc[0][nt] = wmma_bf16(a0.lo, bh, acc[0][nt]);

                acc[1][nt] = wmma_bf16(a1.hi, bh, acc[1][nt]);
                acc[1][nt] = wmma_bf16(a1.hi, bl, acc[1][nt]);
                acc[1][nt] = wmma_bf16(a1.lo, bh, acc[1][nt]);
            }
        }

        // ---- Epilogue: bias + store (C/D layout: VGPR d -> M=d(+8), lane%16 -> N) ----
#pragma unroll
        for (int nt = 0; nt < 4; ++nt) {
            const int   n    = nt * 16 + lrow;
            const float bias = b[n];
#pragma unroll
            for (int mt = 0; mt < 2; ++mt) {
#pragma unroll
                for (int d = 0; d < 8; ++d) {
                    const int row = m0 + mt * 16 + khalf * 8 + d;
                    __builtin_nontemporal_store(acc[mt][nt][d] + bias,
                                                out + (size_t)row * OUT_CH + n);
                }
            }
        }
    }
}

extern "C" void kernel_launch(void* const* d_in, const int* in_sizes, int n_in,
                              void* d_out, int out_size, void* d_ws, size_t ws_size,
                              hipStream_t stream) {
    const float* x = (const float*)d_in[0];
    const float* W = (const float*)d_in[1];
    const float* b = (const float*)d_in[2];
    float* out     = (float*)d_out;

    const size_t lds_bytes = (size_t)LDS_U16 * sizeof(unsigned short); // ~258 KB (<320 KB WGP LDS)
    dim3 grid(256), block(256); // 256 rows per block-iteration -> exactly covers BATCH
    hipLaunchKernelGGL(seplinear_wmma_kernel, grid, block, lds_bytes, stream,
                       x, W, b, out);
}
